// QuantizedLlamaMLP_78151224918032
// MI455X (gfx1250) — 8-bit matrix operand order repaired and checked by reading, compiled, not run on hardware
//
#include <hip/hip_runtime.h>
#include <hip/hip_bf16.h>
#include <stdint.h>

typedef __attribute__((ext_vector_type(8))) int v8i;
typedef __attribute__((ext_vector_type(4))) int v4i;
typedef __attribute__((ext_vector_type(2))) int v2i;

#define QMAX 127.0f
#define EPSQ 1e-8f

#define M_TOK 256
#define H_DIM 4096
#define I_DIM 11008


__global__ void k_zero_scalars(unsigned int* p) {
    if (threadIdx.x < 2) p[threadIdx.x] = 0u;
}

__global__ void k_absmax(const float* __restrict__ x, size_t n, unsigned int* __restrict__ out) {
    __shared__ float red[256];
    float m = 0.0f;
    for (size_t i = (size_t)blockIdx.x * blockDim.x + threadIdx.x; i < n;
         i += (size_t)gridDim.x * blockDim.x)
        m = fmaxf(m, fabsf(x[i]));
    red[threadIdx.x] = m;
    __syncthreads();
    for (int s = 128; s > 0; s >>= 1) {
        if ((int)threadIdx.x < s) red[threadIdx.x] = fmaxf(red[threadIdx.x], red[threadIdx.x + s]);
        __syncthreads();
    }
    if (threadIdx.x == 0) atomicMax(out, __float_as_uint(red[0]));
}

__global__ void k_quant_tensor(const float* __restrict__ x, size_t n,
                               const unsigned int* __restrict__ absbits,
                               int8_t* __restrict__ q) {
    float scale = __uint_as_float(*absbits) / QMAX + EPSQ;
    float inv = 1.0f / scale;
    for (size_t i = (size_t)blockIdx.x * blockDim.x + threadIdx.x; i < n;
         i += (size_t)gridDim.x * blockDim.x) {
        float v = rintf(x[i] * inv);
        v = fminf(fmaxf(v, -128.0f), 127.0f);
        q[i] = (int8_t)v;
    }
}

__global__ void k_quant_rows(const float* __restrict__ w, int cols,
                             int8_t* __restrict__ q, float* __restrict__ scales) {
    __shared__ float red[256];
    __shared__ float s_inv;
    int row = blockIdx.x;
    const float* wr = w + (size_t)row * cols;
    float m = 0.0f;
    for (int c = threadIdx.x; c < cols; c += 256) m = fmaxf(m, fabsf(wr[c]));
    red[threadIdx.x] = m;
    __syncthreads();
    for (int s = 128; s > 0; s >>= 1) {
        if ((int)threadIdx.x < s) red[threadIdx.x] = fmaxf(red[threadIdx.x], red[threadIdx.x + s]);
        __syncthreads();
    }
    if (threadIdx.x == 0) {
        float sc = red[0] / QMAX + EPSQ;
        s_inv = 1.0f / sc;
        scales[row] = sc;
    }
    __syncthreads();
    float inv = s_inv;
    int8_t* qr = q + (size_t)row * cols;
    for (int c = threadIdx.x; c < cols; c += 256) {
        float v = rintf(wr[c] * inv);
        v = fminf(fmaxf(v, -128.0f), 127.0f);
        qr[c] = (int8_t)v;
    }
}


__device__ __forceinline__ v8i load_a_frag(const int8_t* __restrict__ A, int lda,
                                           int m0, int k0, int lane) {
    int l = lane & 15;
    int hi = lane >> 4;
    const int8_t* p = A + (size_t)(m0 + l) * lda + k0 + hi * 16;
    v2i d0 = *(const v2i*)(p);
    v2i d1 = *(const v2i*)(p + 8);
    v2i d2 = *(const v2i*)(p + 32);
    v2i d3 = *(const v2i*)(p + 40);
    v8i r;
    r[0] = d0[0]; r[1] = d0[1];
    r[2] = d1[0]; r[3] = d1[1];
    r[4] = d2[0]; r[5] = d2[1];
    r[6] = d3[0]; r[7] = d3[1];
    return r;
}

__device__ __forceinline__ v8i load_b_frag(const int8_t* __restrict__ W, int ldw,
                                           int c0, int k0, int lane) {
    int n = lane & 15;
    int hi = lane >> 4;
    const int8_t* p = W + (size_t)(c0 + n) * ldw + k0 + hi * 16;
    v4i d0 = *(const v4i*)(p);
    v4i d1 = *(const v4i*)(p + 32);
    v8i r;
    r[0] = d0[0]; r[1] = d0[1]; r[2] = d0[2]; r[3] = d0[3];
    r[4] = d1[0]; r[5] = d1[1]; r[6] = d1[2]; r[7] = d1[3];
    return r;
}

__global__ void __launch_bounds__(128) k_gemm_gate_up(
    const int8_t* __restrict__ xq, const int8_t* __restrict__ wgq,
    const int8_t* __restrict__ wuq,
    const float* __restrict__ sg, const float* __restrict__ su,
    const unsigned int* __restrict__ absx_bits,
    float* __restrict__ h, unsigned int* __restrict__ absh_bits) {
    int lane = threadIdx.x & 31;
    int wave = threadIdx.x >> 5;
    int m0 = blockIdx.y * 32;
    int c0 = (blockIdx.x * 4 + wave) * 64;

    v8i accg0[4] = {0, 0, 0, 0};
    v8i accg1[4] = {0, 0, 0, 0};
    v8i accu0[4] = {0, 0, 0, 0};
    v8i accu1[4] = {0, 0, 0, 0};

    for (int k0 = 0; k0 < H_DIM; k0 += 64) {
        v8i a0 = load_a_frag(xq, H_DIM, m0, k0, lane);
        v8i a1 = load_a_frag(xq, H_DIM, m0 + 16, k0, lane);
#pragma unroll
        for (int t = 0; t < 4; ++t) {
            v8i bg = load_b_frag(wgq, H_DIM, c0 + t * 16, k0, lane);
            accg0[t] = __builtin_amdgcn_wmma_i32_16x16x64_iu8(true, a0, true, bg, accg0[t], false, false);
            accg1[t] = __builtin_amdgcn_wmma_i32_16x16x64_iu8(true, a1, true, bg, accg1[t], false, false);
            v8i bu = load_b_frag(wuq, H_DIM, c0 + t * 16, k0, lane);
            accu0[t] = __builtin_amdgcn_wmma_i32_16x16x64_iu8(true, a0, true, bu, accu0[t], false, false);
            accu1[t] = __builtin_amdgcn_wmma_i32_16x16x64_iu8(true, a1, true, bu, accu1[t], false, false);
        }
    }

    float sx = __uint_as_float(*absx_bits) / QMAX + EPSQ;
    int n = lane & 15;
    int mb = (lane >> 4) * 8;
    float local_max = 0.0f;
#pragma unroll
    for (int t = 0; t < 4; ++t) {
        int c = c0 + t * 16 + n;
        float fg = sx * sg[c];
        float fu = sx * su[c];
#pragma unroll
        for (int r = 0; r < 8; ++r) {
            float g0 = fg * (float)accg0[t][r];
            float u0 = fu * (float)accu0[t][r];
            float hv0 = (g0 / (1.0f + __expf(-g0))) * u0;
            h[(size_t)(m0 + mb + r) * I_DIM + c] = hv0;
            float g1 = fg * (float)accg1[t][r];
            float u1 = fu * (float)accu1[t][r];
            float hv1 = (g1 / (1.0f + __expf(-g1))) * u1;
            h[(size_t)(m0 + 16 + mb + r) * I_DIM + c] = hv1;
            local_max = fmaxf(local_max, fmaxf(fabsf(hv0), fabsf(hv1)));
        }
    }
#pragma unroll
    for (int off = 16; off > 0; off >>= 1)
        local_max = fmaxf(local_max, __shfl_xor(local_max, off, 32));
    if (lane == 0) atomicMax(absh_bits, __float_as_uint(local_max));
}

__global__ void __launch_bounds__(128) k_gemm_down(
    const int8_t* __restrict__ hq, const int8_t* __restrict__ wdq,
    const float* __restrict__ sd, const unsigned int* __restrict__ absh_bits,
    float* __restrict__ out) {
    int lane = threadIdx.x & 31;
    int wave = threadIdx.x >> 5;
    int m0 = blockIdx.y * 32;
    int c0 = (blockIdx.x * 4 + wave) * 64;

    v8i acc0[4] = {0, 0, 0, 0};
    v8i acc1[4] = {0, 0, 0, 0};

    for (int k0 = 0; k0 < I_DIM; k0 += 64) {
        v8i a0 = load_a_frag(hq, I_DIM, m0, k0, lane);
        v8i a1 = load_a_frag(hq, I_DIM, m0 + 16, k0, lane);
#pragma unroll
        for (int t = 0; t < 4; ++t) {
            v8i b = load_b_frag(wdq, I_DIM, c0 + t * 16, k0, lane);
            acc0[t] = __builtin_amdgcn_wmma_i32_16x16x64_iu8(true, a0, true, b, acc0[t], false, false);
            acc1[t] = __builtin_amdgcn_wmma_i32_16x16x64_iu8(true, a1, true, b, acc1[t], false, false);
        }
    }

    float sh = __uint_as_float(*absh_bits) / QMAX + EPSQ;
    int n = lane & 15;
    int mb = (lane >> 4) * 8;
#pragma unroll
    for (int t = 0; t < 4; ++t) {
        int c = c0 + t * 16 + n;
        float f = sh * sd[c];
#pragma unroll
        for (int r = 0; r < 8; ++r) {
            out[(size_t)(m0 + mb + r) * H_DIM + c] = f * (float)acc0[t][r];
            out[(size_t)(m0 + 16 + mb + r) * H_DIM + c] = f * (float)acc1[t][r];
        }
    }
}

extern "C" void kernel_launch(void* const* d_in, const int* in_sizes, int n_in,
                              void* d_out, int out_size, void* d_ws, size_t ws_size,
                              hipStream_t stream) {
    (void)in_sizes; (void)n_in; (void)out_size; (void)ws_size;
    const float* x  = (const float*)d_in[0];
    const float* wg = (const float*)d_in[1];
    const float* wu = (const float*)d_in[2];
    const float* wd = (const float*)d_in[3];
    float* out = (float*)d_out;

    char* ws = (char*)d_ws;
    size_t off = 0;
    auto carve = [&](size_t bytes) -> char* {
        char* p = ws + off;
        off = (off + bytes + 255) & ~(size_t)255;
        return p;
    };

    unsigned int* scal = (unsigned int*)carve(8);
    float* sg  = (float*)carve((size_t)I_DIM * 4);
    float* su  = (float*)carve((size_t)I_DIM * 4);
    float* sd  = (float*)carve((size_t)H_DIM * 4);
    int8_t* xq = (int8_t*)carve((size_t)M_TOK * H_DIM);
    int8_t* hq = (int8_t*)carve((size_t)M_TOK * I_DIM);
    int8_t* wgq = (int8_t*)carve((size_t)I_DIM * H_DIM);
    int8_t* wuq = (int8_t*)carve((size_t)I_DIM * H_DIM);
    int8_t* wdq = (int8_t*)carve((size_t)H_DIM * I_DIM);
    float* hbuf = (float*)carve((size_t)M_TOK * I_DIM * 4);

    k_zero_scalars<<<1, 32, 0, stream>>>(scal);

    size_t nx = (size_t)M_TOK * H_DIM;
    k_absmax<<<512, 256, 0, stream>>>(x, nx, &scal[0]);

    k_quant_rows<<<I_DIM, 256, 0, stream>>>(wg, H_DIM, wgq, sg);
    k_quant_rows<<<I_DIM, 256, 0, stream>>>(wu, H_DIM, wuq, su);
    k_quant_rows<<<H_DIM, 256, 0, stream>>>(wd, I_DIM, wdq, sd);

    k_quant_tensor<<<1024, 256, 0, stream>>>(x, nx, &scal[0], xq);

    dim3 g1(I_DIM / (4 * 64), M_TOK / 32);
    k_gemm_gate_up<<<g1, 128, 0, stream>>>(xq, wgq, wuq, sg, su, &scal[0], hbuf, &scal[1]);

    size_t nh = (size_t)M_TOK * I_DIM;
    k_quant_tensor<<<2048, 256, 0, stream>>>(hbuf, nh, &scal[1], hq);

    dim3 g2(H_DIM / (4 * 64), M_TOK / 32);
    k_gemm_down<<<g2, 128, 0, stream>>>(hq, wdq, sd, &scal[1], out);
}
